// TransConvBlock_309237645716
// MI455X (gfx1250) — matrix operand order repaired and checked by reading, not compiled, not run on hardware
//
#include <hip/hip_runtime.h>
#include <hip/hip_bf16.h>

typedef __attribute__((ext_vector_type(16))) _Float16 v16h;
typedef __attribute__((ext_vector_type(8)))  _Float16 v8h;
typedef __attribute__((ext_vector_type(8)))  float    v8f;

#define HID 128
#define QKVS 512   // q|k|v|skip concatenated
#define EDIM 10
#define LDS_PITCH 36   // floats; 144B rows -> 16B aligned, bank-shifted halves

// ---------- helpers: monotonic float<->uint key for atomicMax segment-max ----------
__device__ __forceinline__ unsigned f2ord(float f) {
    unsigned u = __float_as_uint(f);
    return (u & 0x80000000u) ? ~u : (u | 0x80000000u);
}
__device__ __forceinline__ float ord2f(unsigned u) {
    unsigned v = (u & 0x80000000u) ? (u & 0x7FFFFFFFu) : ~u;
    return __uint_as_float(v);
}

// ---------- 1) pack: x -> f16, {Wq,Wk,Wv,Wskip} -> wh[512][128] f16, biases -> bias[512] ----------
__global__ __launch_bounds__(256) void pack_kernel(
    const float* __restrict__ x,
    const float* __restrict__ Wq, const float* __restrict__ bq,
    const float* __restrict__ Wk, const float* __restrict__ bk,
    const float* __restrict__ Wv, const float* __restrict__ bv,
    const float* __restrict__ Wskip, const float* __restrict__ bskip,
    _Float16* __restrict__ xh, _Float16* __restrict__ wh, float* __restrict__ bias,
    long long nx)
{
    long long stride = (long long)gridDim.x * blockDim.x;
    long long i0 = (long long)blockIdx.x * blockDim.x + threadIdx.x;
    for (long long t = i0; t < nx; t += stride) xh[t] = (_Float16)x[t];
    for (long long t = i0; t < (long long)QKVS * HID; t += stride) {
        int o = (int)(t >> 7), k = (int)(t & 127);
        float w;
        if      (o < 128) w = Wq[o * HID + k];
        else if (o < 256) w = Wk[(o - 128) * HID + k];
        else if (o < 384) w = Wv[(o - 256) * HID + k];
        else              w = Wskip[(o - 384) * HID + k];
        wh[t] = (_Float16)w;
    }
    for (long long t = i0; t < QKVS; t += stride) {
        int o = (int)t;
        bias[o] = (o < 128) ? bq[o] : (o < 256) ? bk[o - 128]
                : (o < 384) ? bv[o - 256] : bskip[o - 384];
    }
}

// ---------- 2) zero/init scratch accumulators ----------
__global__ __launch_bounds__(256) void init_kernel(
    unsigned* __restrict__ amaxk, float* __restrict__ denom,
    float* __restrict__ agg, float* __restrict__ stats, int N)
{
    long long stride = (long long)gridDim.x * blockDim.x;
    long long total = (long long)N * HID;
    for (long long t = (long long)blockIdx.x * blockDim.x + threadIdx.x; t < total; t += stride) {
        agg[t] = 0.0f;
        if (t < N) { amaxk[t] = 0u; denom[t] = 0.0f; }
        if (t < 256) stats[t] = 0.0f;
    }
}

// ---------- 3) fused WMMA GEMM: qkvs[N][512] = x[N][128] @ wh[512][128]^T + bias ----------
// grid = (N/16, 2), block = 256 (8 waves); wave w computes a 16x32 tile:
// otiles (blockIdx.y*8+w)*2 and +1. A fragment reused across both B tiles.
__global__ __launch_bounds__(256) void gemm_qkvs_kernel(
    const _Float16* __restrict__ xh, const _Float16* __restrict__ wh,
    const float* __restrict__ bias, float* __restrict__ out, int N)
{
    __shared__ float lds[8 * 16 * LDS_PITCH];

    const int lane   = threadIdx.x & 31;
    const int wave   = threadIdx.x >> 5;
    const int mtile  = blockIdx.x;
    const int otile0 = (blockIdx.y * 8 + wave) * 2;     // 0,2,...,30
    const int mrow   = mtile * 16 + (lane & 15);
    const int row    = (mrow < N) ? mrow : (N - 1);     // clamp for ragged tail
    const int khalf  = (lane >> 4) * 8;                 // A: K sub-offset 0 or 8
    const int kb     = (lane >> 4) * 16;                // unused: B loads use khalf, same K order as A
    const int ncol0  = otile0 * 16 + (lane & 15);       // B col, tile 0
    const int ncol1  = ncol0 + 16;                      // B col, tile 1

    const _Float16* xrow = xh + (long long)row * HID;
    const _Float16* w0   = wh + (long long)ncol0 * HID;
    const _Float16* w1   = wh + (long long)ncol1 * HID;

    v8f acc0 = {}, acc1 = {};
#pragma unroll
    for (int ks = 0; ks < HID; ks += 32) {
        v8h a_lo = *(const v8h*)(xrow + ks + khalf);
        v8h a_hi = *(const v8h*)(xrow + ks + 16 + khalf);
        v16h a = __builtin_shufflevector(a_lo, a_hi,
                 0, 1, 2, 3, 4, 5, 6, 7, 8, 9, 10, 11, 12, 13, 14, 15);
        v16h b0 = __builtin_shufflevector(*(const v8h*)(w0 + ks + khalf),
                  *(const v8h*)(w0 + ks + 16 + khalf),
                  0, 1, 2, 3, 4, 5, 6, 7, 8, 9, 10, 11, 12, 13, 14, 15);
        v16h b1 = __builtin_shufflevector(*(const v8h*)(w1 + ks + khalf),
                  *(const v8h*)(w1 + ks + 16 + khalf),
                  0, 1, 2, 3, 4, 5, 6, 7, 8, 9, 10, 11, 12, 13, 14, 15);
        acc0 = __builtin_amdgcn_wmma_f32_16x16x32_f16(
                   false, a, false, b0, (short)0, acc0, false, false);
        acc1 = __builtin_amdgcn_wmma_f32_16x16x32_f16(
                   false, a, false, b1, (short)0, acc1, false, false);
    }

    // ---- epilogue: intra-wave LDS transpose -> coalesced b128 stores ----
    float* wl = lds + wave * (16 * LDS_PITCH);
    const int mbase = (lane >> 4) * 8;
    const int nn = lane & 15;
#pragma unroll
    for (int r = 0; r < 8; ++r) {
        wl[(mbase + r) * LDS_PITCH + nn]      = acc0[r];
        wl[(mbase + r) * LDS_PITCH + 16 + nn] = acc1[r];
    }
    // same-wave DS ops are in-order: reads below see the writes above
    const int m  = lane >> 1;             // output row within tile
    const int cb = (lane & 1) * 16;       // column half
    if (mtile * 16 + m < N) {
        const int gcol = otile0 * 16 + cb;
        float* orow = out + (long long)(mtile * 16 + m) * QKVS + gcol;
        const float* bp = bias + gcol;
        const float* lp = wl + m * LDS_PITCH + cb;
#pragma unroll
        for (int j = 0; j < 16; j += 4) {
            float4 v  = *(const float4*)(lp + j);      // ds_load_b128
            float4 bb = *(const float4*)(bp + j);
            v.x += bb.x; v.y += bb.y; v.z += bb.z; v.w += bb.w;
            *(float4*)(orow + j) = v;                  // global_store_b128
        }
    }
}

// ---------- 4) per-edge alpha + segment max (one wave per edge) ----------
__global__ __launch_bounds__(256) void edge_alpha_kernel(
    const int* __restrict__ src, const int* __restrict__ dst,
    const float* __restrict__ ea, const float* __restrict__ We,
    const float* __restrict__ qkvs, float* __restrict__ alpha,
    unsigned* __restrict__ amaxk, int E)
{
    const int lane = threadIdx.x & 31;
    const int nwaves = gridDim.x * (blockDim.x >> 5);
    int wid = blockIdx.x * (blockDim.x >> 5) + (threadIdx.x >> 5);
    for (int e = wid; e < E; e += nwaves) {
        if (e + nwaves < E)
            __builtin_prefetch(ea + (long long)(e + nwaves) * EDIM, 0, 0);
        const int s = src[e], d = dst[e];
        const float* ep = ea + (long long)e * EDIM;
        float ev[EDIM];
#pragma unroll
        for (int j = 0; j < EDIM; ++j) ev[j] = ep[j];
        const float* qrow = qkvs + (long long)d * QKVS;            // q block
        const float* krow = qkvs + (long long)s * QKVS + HID;      // k block
        float part = 0.0f;
#pragma unroll
        for (int c = 0; c < 4; ++c) {
            int dd = lane + c * 32;
            const float* wr = We + dd * EDIM;
            float ed = 0.0f;
#pragma unroll
            for (int j = 0; j < EDIM; ++j) ed += ev[j] * wr[j];
            part += qrow[dd] * (krow[dd] + ed);
        }
#pragma unroll
        for (int off = 16; off; off >>= 1) part += __shfl_xor(part, off, 32);
        if (lane == 0) {
            float a = part * 0.08838834764831845f;   // 1/sqrt(128)
            alpha[e] = a;
            atomicMax(amaxk + d, f2ord(a));
        }
    }
}

// ---------- 5) exp(alpha - max) and segment denominator ----------
__global__ __launch_bounds__(256) void edge_exp_kernel(
    const int* __restrict__ dst, float* __restrict__ alpha,
    const unsigned* __restrict__ amaxk, float* __restrict__ denom, int E)
{
    int stride = gridDim.x * blockDim.x;
    for (int e = blockIdx.x * blockDim.x + threadIdx.x; e < E; e += stride) {
        int d = dst[e];
        float m = ord2f(amaxk[d]);
        float ex = __expf(alpha[e] - m);
        alpha[e] = ex;
        atomicAdd(denom + d, ex);
    }
}

// ---------- 6) weighted scatter-add of (v[src]+e) (one wave per edge) ----------
__global__ __launch_bounds__(256) void edge_agg_kernel(
    const int* __restrict__ src, const int* __restrict__ dst,
    const float* __restrict__ ea, const float* __restrict__ We,
    const float* __restrict__ qkvs, const float* __restrict__ alpha,
    const float* __restrict__ denom, float* __restrict__ agg, int E)
{
    const int lane = threadIdx.x & 31;
    const int nwaves = gridDim.x * (blockDim.x >> 5);
    int wid = blockIdx.x * (blockDim.x >> 5) + (threadIdx.x >> 5);
    for (int e = wid; e < E; e += nwaves) {
        if (e + nwaves < E)
            __builtin_prefetch(ea + (long long)(e + nwaves) * EDIM, 0, 0);
        const int s = src[e], d = dst[e];
        const float w = alpha[e] / denom[d];
        const float* ep = ea + (long long)e * EDIM;
        float ev[EDIM];
#pragma unroll
        for (int j = 0; j < EDIM; ++j) ev[j] = ep[j];
        const float* vrow = qkvs + (long long)s * QKVS + 2 * HID;  // v block
        float* arow = agg + (long long)d * HID;
#pragma unroll
        for (int c = 0; c < 4; ++c) {
            int dd = lane + c * 32;
            const float* wr = We + dd * EDIM;
            float ed = 0.0f;
#pragma unroll
            for (int j = 0; j < EDIM; ++j) ed += ev[j] * wr[j];
            atomicAdd(arow + dd, w * (vrow[dd] + ed));
        }
    }
}

// ---------- 7) out = relu(agg + skip); channel sums for BN (block partials) ----------
__global__ __launch_bounds__(128) void relu_stats_kernel(
    const float* __restrict__ agg, const float* __restrict__ qkvs,
    float* __restrict__ h, float* __restrict__ stats, int N, int rowsPerBlock)
{
    const int c = threadIdx.x;                 // channel
    int r0 = blockIdx.x * rowsPerBlock;
    int r1 = r0 + rowsPerBlock; if (r1 > N) r1 = N;
    float s1 = 0.0f, s2 = 0.0f;
    for (int n = r0; n < r1; ++n) {
        float v = agg[(long long)n * HID + c] + qkvs[(long long)n * QKVS + 3 * HID + c];
        v = fmaxf(v, 0.0f);
        h[(long long)n * HID + c] = v;
        s1 += v; s2 += v * v;
    }
    atomicAdd(&stats[c], s1);
    atomicAdd(&stats[HID + c], s2);
}

// ---------- 8) BatchNorm (training stats, biased var) ----------
__global__ __launch_bounds__(256) void bn_kernel(
    float* __restrict__ h, const float* __restrict__ stats,
    const float* __restrict__ gamma, const float* __restrict__ beta, int N)
{
    long long total = (long long)N * HID;
    long long stride = (long long)gridDim.x * blockDim.x;
    const float invN = 1.0f / (float)N;
    for (long long t = (long long)blockIdx.x * blockDim.x + threadIdx.x; t < total; t += stride) {
        int c = (int)(t & 127);
        float mean = stats[c] * invN;
        float var  = stats[HID + c] * invN - mean * mean;
        h[t] = gamma[c] * (h[t] - mean) * rsqrtf(var + 1e-5f) + beta[c];
    }
}

extern "C" void kernel_launch(void* const* d_in, const int* in_sizes, int n_in,
                              void* d_out, int out_size, void* d_ws, size_t ws_size,
                              hipStream_t stream)
{
    const float* x        = (const float*)d_in[0];
    const int*   eidx     = (const int*)d_in[1];
    const float* eattr    = (const float*)d_in[2];
    const float* Wq       = (const float*)d_in[3];
    const float* bq       = (const float*)d_in[4];
    const float* Wk       = (const float*)d_in[5];
    const float* bk       = (const float*)d_in[6];
    const float* Wv       = (const float*)d_in[7];
    const float* bv       = (const float*)d_in[8];
    const float* We       = (const float*)d_in[9];
    const float* Wskip    = (const float*)d_in[10];
    const float* bskip    = (const float*)d_in[11];
    const float* gamma    = (const float*)d_in[12];
    const float* beta     = (const float*)d_in[13];

    const int N = in_sizes[0] / HID;
    const int E = in_sizes[1] / 2;
    const int* src = eidx;
    const int* dst = eidx + E;

    // workspace carve-up (256B aligned)
    char* p = (char*)d_ws;
    auto carve = [&](size_t bytes) { char* r = p; p += (bytes + 255) & ~(size_t)255; return r; };
    _Float16* xh    = (_Float16*)carve((size_t)N * HID * sizeof(_Float16));
    _Float16* wh    = (_Float16*)carve((size_t)QKVS * HID * sizeof(_Float16));
    float*    bias  = (float*)carve(QKVS * sizeof(float));
    float*    qkvs  = (float*)carve((size_t)N * QKVS * sizeof(float));
    float*    alpha = (float*)carve((size_t)E * sizeof(float));
    unsigned* amaxk = (unsigned*)carve((size_t)N * sizeof(unsigned));
    float*    denom = (float*)carve((size_t)N * sizeof(float));
    float*    agg   = (float*)carve((size_t)N * HID * sizeof(float));
    float*    stats = (float*)carve(256 * sizeof(float));

    float* out = (float*)d_out;

    // 1) pack / convert
    pack_kernel<<<2048, 256, 0, stream>>>(x, Wq, bq, Wk, bk, Wv, bv, Wskip, bskip,
                                          xh, wh, bias, (long long)N * HID);
    // 2) init accumulators
    init_kernel<<<2048, 256, 0, stream>>>(amaxk, denom, agg, stats, N);
    // 3) fused q|k|v|skip GEMM via WMMA (16x32 tile per wave)
    dim3 ggrid((N + 15) / 16, 2, 1);
    gemm_qkvs_kernel<<<ggrid, 256, 0, stream>>>(xh, wh, bias, qkvs, N);
    // 4) attention logits + segment max
    int wblocks = (E + 7) / 8; if (wblocks > 4096) wblocks = 4096; if (wblocks < 1) wblocks = 1;
    edge_alpha_kernel<<<wblocks, 256, 0, stream>>>(src, dst, eattr, We, qkvs, alpha, amaxk, E);
    // 5) exp + denom
    int tblocks = (E + 255) / 256; if (tblocks > 4096) tblocks = 4096; if (tblocks < 1) tblocks = 1;
    edge_exp_kernel<<<tblocks, 256, 0, stream>>>(dst, alpha, amaxk, denom, E);
    // 6) weighted aggregation scatter
    edge_agg_kernel<<<wblocks, 256, 0, stream>>>(src, dst, eattr, We, qkvs, alpha, denom, agg, E);
    // 7) skip + relu + BN stats
    int nblocks = 512;
    int rowsPerBlock = (N + nblocks - 1) / nblocks;
    relu_stats_kernel<<<nblocks, 128, 0, stream>>>(agg, qkvs, out, stats, N, rowsPerBlock);
    // 8) normalize
    bn_kernel<<<2048, 256, 0, stream>>>(out, stats, gamma, beta, N);
}
